// Attention_23854248362142
// MI455X (gfx1250) — matrix operand order repaired and checked by reading, not compiled, not run on hardware
//
#include <hip/hip_runtime.h>

// ---------------------------------------------------------------------------
// MHA forward for B=2,S=2048,D=1024,H=16 on gfx1250 (MI455X).
// Compute-bound (~69 GFLOP vs ~50MB traffic) -> all matmuls through
// v_wmma_f32_16x16x32_bf16 (fp32 accum); plain-copy tile loads through the
// Tensor Data Mover (tensor_load_to_lds + s_wait_tensorcnt).
// ---------------------------------------------------------------------------

typedef __attribute__((ext_vector_type(16))) __bf16 v16bf;
typedef __attribute__((ext_vector_type(8)))  float  v8f;
typedef __attribute__((ext_vector_type(4)))  unsigned int u32x4;
typedef __attribute__((ext_vector_type(8)))  int i32x8;
typedef __attribute__((ext_vector_type(4)))  int i32x4;

#define DEV static __device__ __forceinline__

DEV unsigned short f32_to_bf16_rne(float f) {
  unsigned int u = __builtin_bit_cast(unsigned int, f);
  u += 0x7FFFu + ((u >> 16) & 1u);   // round-to-nearest-even
  return (unsigned short)(u >> 16);
}
DEV __bf16 to_bf16(float f) {
  union { unsigned short s; __bf16 b; } u;
  u.s = f32_to_bf16_rne(f);
  return u.b;
}

// A-matrix 16x32 bf16 fragment: lane idx=row, per-lane k chunks {g*8.., 16+g*8..}
DEV v16bf load_frag_a(const __bf16* rowptr, int grp) {
  union { uint4 u[2]; v16bf v; } f;
  f.u[0] = *reinterpret_cast<const uint4*>(rowptr + grp * 8);
  f.u[1] = *reinterpret_cast<const uint4*>(rowptr + 16 + grp * 8);
  return f.v;
}
// B-matrix 32x16 bf16 fragment: lane idx=col, per-lane k chunks {g*8.., 16+g*8..}
DEV v16bf load_frag_b(const __bf16* rowptr, int grp) {
  union { uint4 u[2]; v16bf v; } f;
  f.u[0] = *reinterpret_cast<const uint4*>(rowptr + grp * 8);
  f.u[1] = *reinterpret_cast<const uint4*>(rowptr + 16 + grp * 8);
  return f.v;
}

// ---------------------------------------------------------------------------
// TDM: 2-D tile DMA global -> LDS, bf16 elements, optional LDS row padding.
// D# bitfields per CDNA5 ISA.
//   pad: after every (8 << pad_itv) bytes, skip (pad_amt+1) dwords in LDS.
// ---------------------------------------------------------------------------
DEV void tdm_load_2d_bf16(unsigned lds_off, const void* gptr,
                          unsigned tile_x, unsigned tile_y,
                          unsigned tensor_x, unsigned tensor_y,
                          unsigned stride_x,
                          unsigned pad_itv, unsigned pad_amt, unsigned pad_en) {
  unsigned long long ga = (unsigned long long)(size_t)gptr;
  u32x4 g0;
  g0.x = 1u;                                   // count=1, user descriptor
  g0.y = lds_off;                              // lds_addr (bytes)
  g0.z = (unsigned)ga;                         // global_addr[31:0]
  g0.w = (unsigned)(ga >> 32) | (2u << 30);    // global_addr[56:32] | type=2
  i32x8 g1;
  g1[0] = (int)((1u << 16) | (pad_en << 20) | (pad_itv << 22) | (pad_amt << 25));
  g1[1] = (int)(tensor_x << 16);                              // dim0 lo16
  g1[2] = (int)((tensor_x >> 16) | (tensor_y << 16));         // dim0 hi | dim1 lo
  g1[3] = (int)((tensor_y >> 16) | (tile_x << 16));           // dim1 hi | tile0
  g1[4] = (int)tile_y;                                        // tile1 | tile2=0
  g1[5] = (int)stride_x;                                      // dim0_stride lo32
  g1[6] = 0;
  g1[7] = 0;
  i32x4 z4 = {0, 0, 0, 0};
#if defined(__clang_major__) && (__clang_major__ >= 23)
  i32x8 z8 = {0, 0, 0, 0, 0, 0, 0, 0};
  __builtin_amdgcn_tensor_load_to_lds(g0, g1, z4, z4, z8, 0);
#else
  __builtin_amdgcn_tensor_load_to_lds(g0, g1, z4, z4, 0);
#endif
}

// ---------------------------------------------------------------------------
// fp32 -> bf16 conversion (grid-strided)
// ---------------------------------------------------------------------------
__global__ __launch_bounds__(256)
void cvt_f32_bf16_kernel(const float* __restrict__ in,
                         __bf16* __restrict__ out, int n) {
  int i = blockIdx.x * 256 + threadIdx.x;
  const int stride = gridDim.x * 256;
  for (; i < n; i += stride)
    reinterpret_cast<unsigned short*>(out)[i] = f32_to_bf16_rne(in[i]);
}

// ---------------------------------------------------------------------------
// C[M,N] = A[M,K]_bf16 @ W[K,N]_bf16 + bias[N], 128x128 tile / block,
// 8 waves, each wave 64x32 (4x2 WMMA accumulators). OUT_F32 picks epilogue.
// A tile arrives via TDM (wave 0), W tile manual transpose (all threads).
// ---------------------------------------------------------------------------
template <bool OUT_F32>
__global__ __launch_bounds__(256)
void gemm_bias_kernel(const __bf16* __restrict__ A,
                      const __bf16* __restrict__ W,
                      const float*  __restrict__ bias,
                      void* __restrict__ Cout,
                      int M, int N, int K) {
  constexpr int TS = 128, KS = 32, ST = 56;  // ST: padded LDS row stride (bf16)
  __shared__ __bf16 sA[TS * ST];             // [row][k]   (TDM writes w/ pad)
  __shared__ __bf16 sB[TS * ST];             // [col][k]   (W transposed on load)

  const int tid = threadIdx.x, lane = tid & 31, wave = tid >> 5;
  const int wm = wave >> 2, wn = wave & 3;
  const int idx = lane & 15, grp = lane >> 4;
  const int m0 = blockIdx.x * TS, n0 = blockIdx.y * TS;

  v8f acc[4][2] = {};

  const int wr = tid >> 3, wc = (tid & 7) * 16;   // W-tile loader: 32x128

  for (int k0 = 0; k0 < K; k0 += KS) {
    if (wave == 0) {
      // A tile 128 rows x 32 k: row = 64B -> pad_itv=3 (8<<3),
      // +12 dwords (pad_amt=11) -> LDS row stride 56 bf16 == ST.
      tdm_load_2d_bf16((unsigned)(size_t)&sA[0],
                       A + (size_t)m0 * K + k0,
                       /*tile_x=*/KS, /*tile_y=*/TS,
                       /*tensor_x=*/KS, /*tensor_y=*/TS,
                       /*stride_x=*/(unsigned)K,
                       /*pad_itv=*/3, /*pad_amt=*/11, /*pad_en=*/1);
      __builtin_amdgcn_s_wait_tensorcnt((unsigned short)0);
    }
    {  // W tile: coalesced rows, transposed scatter into sB[col][k]
      union { uint4 u[2]; __bf16 e[16]; } d;
      const uint4* g =
          reinterpret_cast<const uint4*>(W + (size_t)(k0 + wr) * N + n0 + wc);
      d.u[0] = g[0]; d.u[1] = g[1];
#pragma unroll
      for (int j = 0; j < 16; ++j) sB[(wc + j) * ST + wr] = d.e[j];
    }
    if (k0 + KS < K)
      __builtin_prefetch(W + (size_t)(k0 + KS + wr) * N + n0 + wc, 0, 1);
    __syncthreads();

    v16bf bf[2];
#pragma unroll
    for (int ni = 0; ni < 2; ++ni)
      bf[ni] = load_frag_b(&sB[(wn * 32 + ni * 16 + idx) * ST], grp);
#pragma unroll
    for (int mi = 0; mi < 4; ++mi) {
      v16bf af = load_frag_a(&sA[(wm * 64 + mi * 16 + idx) * ST], grp);
#pragma unroll
      for (int ni = 0; ni < 2; ++ni)
        acc[mi][ni] = __builtin_amdgcn_wmma_f32_16x16x32_bf16(
            false, af, false, bf[ni], (short)0, acc[mi][ni], false, false);
    }
    __syncthreads();
  }

  // epilogue: C/D layout row = v + 8*grp, col = idx
#pragma unroll
  for (int ni = 0; ni < 2; ++ni) {
    const int nc = n0 + wn * 32 + ni * 16 + idx;
    const float bv = bias[nc];
#pragma unroll
    for (int mi = 0; mi < 4; ++mi) {
      const int mrb = m0 + wm * 64 + mi * 16 + grp * 8;
#pragma unroll
      for (int v = 0; v < 8; ++v) {
        const float val = acc[mi][ni][v] + bv;
        if (OUT_F32)
          reinterpret_cast<float*>(Cout)[(size_t)(mrb + v) * N + nc] = val;
        else
          reinterpret_cast<unsigned short*>(Cout)[(size_t)(mrb + v) * N + nc] =
              f32_to_bf16_rne(val);
      }
    }
  }
}

// ---------------------------------------------------------------------------
// Flash attention: one block per (b,h, 128-query tile). 8 waves x 16 rows.
// K tile via TDM, V tile manual transpose, online softmax, WMMA everywhere.
// ---------------------------------------------------------------------------
__global__ __launch_bounds__(256)
void flash_attn_kernel(const __bf16* __restrict__ Q,
                       const __bf16* __restrict__ Kmat,
                       const __bf16* __restrict__ Vmat,
                       __bf16* __restrict__ O) {
  constexpr int S = 2048, Dm = 1024, DEPTH = 64, KT = 64;
  constexpr int SK = 72, SV = 72, SP = 72;    // padded strides (mult of 8 bf16)
  __shared__ __bf16 sK[KT * SK];              // [key][d]   (TDM writes w/ pad)
  __shared__ __bf16 sVt[DEPTH * SV];          // [d][key]   (transposed)
  __shared__ __bf16 sP[8 * 16 * SP];          // per-wave P strip [16][KT]

  const int tid = threadIdx.x, lane = tid & 31, wave = tid >> 5;
  const int idx = lane & 15, grp = lane >> 4;
  const int bh = blockIdx.y, b = bh >> 4, h = bh & 15;
  const int q0 = blockIdx.x * 128;
  const size_t base = (size_t)b * S * Dm + (size_t)h * DEPTH;

  // Q fragments (16 rows x 64 depth = two 16x32 A-fragments), straight from L2
  const __bf16* qrow = Q + base + (size_t)(q0 + wave * 16 + idx) * Dm;
  v16bf qf[2];
#pragma unroll
  for (int f = 0; f < 2; ++f) {
    union { uint4 u[2]; v16bf v; } u;
    u.u[0] = *reinterpret_cast<const uint4*>(qrow + f * 32 + grp * 8);
    u.u[1] = *reinterpret_cast<const uint4*>(qrow + f * 32 + 16 + grp * 8);
    qf[f] = u.v;
  }

  float m_run[8], l_run[8];
  v8f acc[4] = {};
#pragma unroll
  for (int v = 0; v < 8; ++v) { m_run[v] = -__builtin_inff(); l_run[v] = 0.f; }

  const float C = 0.125f * 1.44269504088896340736f;  // 1/sqrt(64) * log2(e)
  const int lr = tid >> 2, ld = (tid & 3) * 16;      // V-tile loader indices

  for (int kt = 0; kt < S; kt += KT) {
    if (wave == 0) {
      // K tile 64 keys x 64 d: row = 128B -> pad_itv=4 (8<<4),
      // +4 dwords (pad_amt=3) -> LDS row stride 72 bf16 == SK.
      tdm_load_2d_bf16((unsigned)(size_t)&sK[0],
                       Kmat + base + (size_t)kt * Dm,
                       /*tile_x=*/DEPTH, /*tile_y=*/KT,
                       /*tensor_x=*/DEPTH, /*tensor_y=*/KT,
                       /*stride_x=*/Dm,
                       /*pad_itv=*/4, /*pad_amt=*/3, /*pad_en=*/1);
      __builtin_amdgcn_s_wait_tensorcnt((unsigned short)0);
    }
    {  // V tile -> sVt[d][key] (transposed scatter)
      union { uint4 u[2]; __bf16 e[16]; } dv;
      const uint4* g = reinterpret_cast<const uint4*>(
          Vmat + base + (size_t)(kt + lr) * Dm + ld);
      dv.u[0] = g[0]; dv.u[1] = g[1];
#pragma unroll
      for (int j = 0; j < 16; ++j) sVt[(ld + j) * SV + lr] = dv.e[j];
    }
    if (kt + KT < S)  // pull next V tile toward L2 (global_prefetch_b8)
      __builtin_prefetch(Vmat + base + (size_t)(kt + KT + lr) * Dm + ld, 0, 1);
    __syncthreads();

    // scores: 4 key blocks of 16, K-dim = depth (2 x 32)
    v8f sb[4];
#pragma unroll
    for (int blk = 0; blk < 4; ++blk) {
      v8f s = {};
#pragma unroll
      for (int ks = 0; ks < 2; ++ks) {
        v16bf kf = load_frag_b(&sK[(blk * 16 + idx) * SK + ks * 32], grp);
        s = __builtin_amdgcn_wmma_f32_16x16x32_bf16(
            false, qf[ks], false, kf, (short)0, s, false, false);
      }
      sb[blk] = s;
    }

    // online softmax; row r = grp*8+v lives in vgpr v across a 16-lane group
    __bf16* pw = sP + wave * 16 * SP;
#pragma unroll
    for (int v = 0; v < 8; ++v) {
      float mx = m_run[v];
#pragma unroll
      for (int blk = 0; blk < 4; ++blk) mx = fmaxf(mx, sb[blk][v]);
#pragma unroll
      for (int mk = 1; mk < 16; mk <<= 1)
        mx = fmaxf(mx, __shfl_xor(mx, mk, 32));
      const float alpha = exp2f((m_run[v] - mx) * C);
      m_run[v] = mx;
      l_run[v] *= alpha;
#pragma unroll
      for (int db = 0; db < 4; ++db) acc[db][v] *= alpha;
      float rs = 0.f;
#pragma unroll
      for (int blk = 0; blk < 4; ++blk) {
        const float p = exp2f((sb[blk][v] - mx) * C);
        rs += p;
        pw[(grp * 8 + v) * SP + blk * 16 + idx] = to_bf16(p);  // C->A reshape
      }
#pragma unroll
      for (int mk = 1; mk < 16; mk <<= 1) rs += __shfl_xor(rs, mk, 32);
      l_run[v] += rs;
    }

    // PV: acc[16 x 64] += P[16 x 64] @ V[64 x 64]  (per-wave LDS, DS in-order)
#pragma unroll
    for (int kb = 0; kb < 2; ++kb) {
      v16bf pf = load_frag_a(&pw[idx * SP + kb * 32], grp);
#pragma unroll
      for (int db = 0; db < 4; ++db) {
        v16bf vf = load_frag_b(&sVt[(db * 16 + idx) * SV + kb * 32], grp);
        acc[db] = __builtin_amdgcn_wmma_f32_16x16x32_bf16(
            false, pf, false, vf, (short)0, acc[db], false, false);
      }
    }
    __syncthreads();
  }

  // normalize and store bf16 context at [b, q, h*64 + d]
#pragma unroll
  for (int v = 0; v < 8; ++v) {
    const float inv = 1.0f / l_run[v];
    const size_t orow = base + (size_t)(q0 + wave * 16 + grp * 8 + v) * Dm;
#pragma unroll
    for (int db = 0; db < 4; ++db)
      reinterpret_cast<unsigned short*>(O)[orow + db * 16 + idx] =
          f32_to_bf16_rne(acc[db][v] * inv);
  }
}

// ---------------------------------------------------------------------------
// Host launcher
// ---------------------------------------------------------------------------
extern "C" void kernel_launch(void* const* d_in, const int* in_sizes, int n_in,
                              void* d_out, int out_size, void* d_ws,
                              size_t ws_size, hipStream_t stream) {
  (void)in_sizes; (void)n_in; (void)out_size; (void)ws_size;
  constexpr int Bb = 2, Ss = 2048, Dd = 1024;
  constexpr int M = Bb * Ss;

  const float* X  = (const float*)d_in[0];
  const float* Wq = (const float*)d_in[1];
  const float* bq = (const float*)d_in[2];
  const float* Wk = (const float*)d_in[3];
  const float* bk = (const float*)d_in[4];
  const float* Wv = (const float*)d_in[5];
  const float* bv = (const float*)d_in[6];
  const float* Wo = (const float*)d_in[7];
  const float* bo = (const float*)d_in[8];

  __bf16* p   = (__bf16*)d_ws;
  __bf16* Xb  = p; p += (size_t)M * Dd;
  __bf16* Wqb = p; p += (size_t)Dd * Dd;
  __bf16* Wkb = p; p += (size_t)Dd * Dd;
  __bf16* Wvb = p; p += (size_t)Dd * Dd;
  __bf16* Wob = p; p += (size_t)Dd * Dd;
  __bf16* Qb  = p; p += (size_t)M * Dd;
  __bf16* Kb  = p; p += (size_t)M * Dd;
  __bf16* Vb  = p; p += (size_t)M * Dd;
  __bf16* Ob  = p; p += (size_t)M * Dd;

  cvt_f32_bf16_kernel<<<2048, 256, 0, stream>>>(X,  Xb,  M * Dd);
  cvt_f32_bf16_kernel<<<1024, 256, 0, stream>>>(Wq, Wqb, Dd * Dd);
  cvt_f32_bf16_kernel<<<1024, 256, 0, stream>>>(Wk, Wkb, Dd * Dd);
  cvt_f32_bf16_kernel<<<1024, 256, 0, stream>>>(Wv, Wvb, Dd * Dd);
  cvt_f32_bf16_kernel<<<1024, 256, 0, stream>>>(Wo, Wob, Dd * Dd);

  dim3 gg(M / 128, Dd / 128);
  gemm_bias_kernel<false><<<gg, 256, 0, stream>>>(Xb, Wqb, bq, Qb, M, Dd, Dd);
  gemm_bias_kernel<false><<<gg, 256, 0, stream>>>(Xb, Wkb, bk, Kb, M, Dd, Dd);
  gemm_bias_kernel<false><<<gg, 256, 0, stream>>>(Xb, Wvb, bv, Vb, M, Dd, Dd);

  dim3 fg(Ss / 128, Bb * 16);
  flash_attn_kernel<<<fg, 256, 0, stream>>>(Qb, Kb, Vb, Ob);

  gemm_bias_kernel<true><<<gg, 256, 0, stream>>>(Ob, Wob, bo, d_out, M, Dd, Dd);
}
